// ChaiThermoTransformer_18803366822086
// MI455X (gfx1250) — compile-verified, matrix operand order checked by reading, not run on hardware
//
#include <hip/hip_runtime.h>
#include <hip/hip_bf16.h>
#include <math.h>

// ---------------------------------------------------------------------------
// CDNA5 (gfx1250) wave32 WMMA implementation of the Chai thermo-transformer.
// - all GEMMs via v_wmma_f32_16x16x32_bf16 (packed-B fragments -> b128 loads)
// - pair tensor (604 MB, the bandwidth roofline) streamed with
//   global_load_async_to_lds_b128 double buffering (ASYNCcnt)
// ---------------------------------------------------------------------------

typedef __attribute__((ext_vector_type(16))) __bf16 v16bf;
typedef __attribute__((ext_vector_type(8)))  __bf16 v8bf;
typedef __attribute__((ext_vector_type(8)))  float  v8f;

#define LSEQ 768
#define DM   256
#define NH   8
#define HD   32
#define DFF  512
#define PAIRC 256
#define MMUT 64
#define PB_GRID 4608

__device__ __forceinline__ v8f wmma_bf16(v16bf a, v16bf b, v8f c) {
  return __builtin_amdgcn_wmma_f32_16x16x32_bf16(
      /*neg_a=*/false, a, /*neg_b=*/false, b,
      /*c_mod=*/(short)0, c, /*reuse_a=*/false, /*reuse_b=*/false);
}

__device__ __forceinline__ v16bf frag_cat(v8bf lo, v8bf hi) {
  return __builtin_shufflevector(lo, hi, 0, 1, 2, 3, 4, 5, 6, 7,
                                         8, 9, 10, 11, 12, 13, 14, 15);
}

// A-fragment (16x32 bf16, row-major source, leading dim lda; lda % 8 == 0).
__device__ __forceinline__ v16bf load_a_bf(const __bf16* base, int lda) {
  int l = threadIdx.x & 31;
  const __bf16* p = base + (l & 15) * lda + ((l & 16) ? 8 : 0);
  v8bf lo = *(const v8bf*)p;
  v8bf hi = *(const v8bf*)(p + 16);
  return frag_cat(lo, hi);
}

// B-fragment from pre-packed weights: lane l reads 16 contiguous bf16.
__device__ __forceinline__ v16bf load_b_packed(const __bf16* frag_base) {
  const __bf16* p = frag_base + (threadIdx.x & 31) * 16;
  v8bf lo = *(const v8bf*)p;
  v8bf hi = *(const v8bf*)(p + 8);
  return frag_cat(lo, hi);
}

// Async global -> LDS copy of 16 bytes per lane (ASYNCcnt-tracked).
__device__ __forceinline__ void async_copy_b128(void* lds_dst, const void* gsrc) {
  unsigned lds_addr = (unsigned)(size_t)lds_dst;             // LDS byte offset
  unsigned long long ga = (unsigned long long)(size_t)gsrc;  // 64-bit vaddr
  asm volatile("global_load_async_to_lds_b128 %0, %1, off"
               :: "v"(lds_addr), "v"(ga) : "memory");
}
__device__ __forceinline__ void async_wait0() {
  asm volatile("s_wait_asynccnt 0x0" ::: "memory");
}

__device__ __forceinline__ float gelu_exact(float x) {
  return 0.5f * x * (1.0f + erff(x * 0.7071067811865475f));
}
__device__ __forceinline__ float hmax16(float x) {
#pragma unroll
  for (int m = 1; m < 16; m <<= 1) x = fmaxf(x, __shfl_xor(x, m, 32));
  return x;
}
__device__ __forceinline__ float hsum16(float x) {
#pragma unroll
  for (int m = 1; m < 16; m <<= 1) x += __shfl_xor(x, m, 32);
  return x;
}

// ---------------------------------------------------------------------------
// Weight packing: fp32 row-major (K x N) -> bf16 WMMA B-fragment order.
// Fragment f = ntile*(K/32)+ktile holds 512 bf16: lane-major, 16 values/lane.
// ---------------------------------------------------------------------------
__global__ void k_packB(const float* __restrict__ W, __bf16* __restrict__ P,
                        int K, int N) {
  int tid = blockIdx.x * 256 + threadIdx.x;
  if (tid >= K * N) return;
  int nK = K >> 5;
  int frag = tid >> 9;
  int within = tid & 511;
  int lane = within >> 4, i = within & 15;
  int ntile = frag / nK, ktile = frag - ntile * nK;
  int n = ntile * 16 + (lane & 15);
  int kb = (lane & 16) ? 8 : 0;
  int k = ktile * 32 + ((i < 8) ? (kb + i) : (16 + kb + (i - 8)));
  P[tid] = (__bf16)W[k * N + n];
}
// pb_w2 (64x8) -> packed fragments of a logical 64x16 (cols 8..15 zero)
__global__ void k_packw2(const float* __restrict__ s, __bf16* __restrict__ d) {
  int tid = blockIdx.x * 256 + threadIdx.x;
  if (tid >= 1024) return;
  int frag = tid >> 9;
  int within = tid & 511;
  int lane = within >> 4, i = within & 15;
  int n = lane & 15;
  int kb = (lane & 16) ? 8 : 0;
  int k = frag * 32 + ((i < 8) ? (kb + i) : (16 + kb + (i - 8)));
  d[tid] = (__bf16)(n < 8 ? s[k * 8 + n] : 0.0f);
}

// ---------------------------------------------------------------------------
// Generic fused [LN ->] GEMM (bf16 WMMA, fp32 accum) [-> act/residual/LN]
// One block = 16 rows. 8 waves, wave w handles column tiles w, w+8, ...
// ---------------------------------------------------------------------------
__global__ __launch_bounds__(256)
void k_gemm(const float* __restrict__ Xf, const __bf16* __restrict__ Xb,
            int K, int N,
            const float* __restrict__ lng, const float* __restrict__ lnb,
            const __bf16* __restrict__ W, const float* __restrict__ bias,
            int act,                                   // 0 none, 1 gelu
            const float* __restrict__ lng2, const float* __restrict__ lnb2,
            int addres,
            float* __restrict__ Yf, __bf16* __restrict__ Yb,
            __bf16* __restrict__ YbT) {
  __shared__ __align__(16) __bf16 sA[16 * 512];
  __shared__ float  sO[16 * 256];   // only used when lng2 != null (N==256)
  __shared__ float  mu_s[16], rs_s[16];
  const int row0 = blockIdx.x * 16;
  const int tid  = threadIdx.x;

  if (Xf) {
    if (lng) {
      if (tid < 16) {
        const float* xr = Xf + (size_t)(row0 + tid) * K;
        float s = 0.f;
        for (int k = 0; k < K; ++k) s += xr[k];
        float mu = s / (float)K, v = 0.f;
        for (int k = 0; k < K; ++k) { float d = xr[k] - mu; v += d * d; }
        mu_s[tid] = mu;
        rs_s[tid] = rsqrtf(v / (float)K + 1e-5f);
      }
      __syncthreads();
      for (int idx = tid; idx < 16 * K; idx += 256) {
        int r = idx / K, k = idx - r * K;
        float x = Xf[(size_t)(row0 + r) * K + k];
        sA[idx] = (__bf16)((x - mu_s[r]) * rs_s[r] * lng[k] + lnb[k]);
      }
    } else {
      for (int idx = tid; idx < 16 * K; idx += 256) {
        int r = idx / K, k = idx - r * K;
        sA[idx] = (__bf16)Xf[(size_t)(row0 + r) * K + k];
      }
    }
  } else {
    for (int idx = tid; idx < 16 * K; idx += 256) {
      int r = idx / K, k = idx - r * K;
      sA[idx] = Xb[(size_t)(row0 + r) * K + k];
    }
  }
  __syncthreads();

  const int wave = tid >> 5;
  const int r0   = (tid & 16) ? 8 : 0;
  const int col  = tid & 15;
  const int ntiles = N >> 4;
  const int nK = K >> 5;

  for (int t = wave; t < ntiles; t += 8) {
    const int n0 = t * 16;
    const __bf16* wt = W + (size_t)t * nK * 512;
    v8f acc = {};
    for (int ks = 0; ks < nK; ++ks) {
      v16bf a = load_a_bf(sA + ks * 32, K);
      v16bf b = load_b_packed(wt + (size_t)ks * 512);
      acc = wmma_bf16(a, b, acc);
    }
#pragma unroll
    for (int v = 0; v < 8; ++v) {
      int r = r0 + v, n = n0 + col;
      float y = acc[v] + (bias ? bias[n] : 0.f);
      if (act == 1) y = gelu_exact(y);
      if (lng2) {
        sO[r * 256 + n] = y;
      } else {
        size_t gi = (size_t)(row0 + r) * N + n;
        if (addres) y += Yf[gi];
        if (Yf) Yf[gi] = y;
        if (Yb) Yb[gi] = (__bf16)y;
        if (YbT && n >= 2 * DM)        // V section, transposed [dim][token]
          YbT[(size_t)(n - 2 * DM) * LSEQ + (row0 + r)] = (__bf16)y;
      }
    }
  }

  if (lng2) {   // output LayerNorm over N==256 (single-proj path)
    __syncthreads();
    if (tid < 16) {
      float s = 0.f;
      for (int n = 0; n < 256; ++n) s += sO[tid * 256 + n];
      float mu = s / 256.f, v = 0.f;
      for (int n = 0; n < 256; ++n) { float d = sO[tid * 256 + n] - mu; v += d * d; }
      mu_s[tid] = mu;
      rs_s[tid] = rsqrtf(v / 256.f + 1e-5f);
    }
    __syncthreads();
    for (int idx = tid; idx < 16 * 256; idx += 256) {
      int r = idx >> 8, n = idx & 255;
      Yf[(size_t)(row0 + r) * 256 + n] =
          (sO[idx] - mu_s[r]) * rs_s[r] * lng2[n] + lnb2[n];
    }
  }
}

// ---------------------------------------------------------------------------
// Pair-bias kernel (dominant; reads 604 MB pair exactly once at the HBM
// roofline). Persistent blocks, double-buffered async global->LDS staging:
// while tile t is LayerNorm'd + WMMA'd from buffer cur, tile t+grid streams
// into buffer cur^1 via global_load_async_to_lds_b128.
// ---------------------------------------------------------------------------
__global__ __launch_bounds__(64)
void k_pairbias(const float* __restrict__ pair,
                const float* __restrict__ lng, const float* __restrict__ lnb,
                const __bf16* __restrict__ w1p, const float* __restrict__ b1,
                const __bf16* __restrict__ w2p, const float* __restrict__ b2,
                const float* __restrict__ bias_scale,
                float* __restrict__ pb_out) {
  __shared__ __align__(16) float  sX[2][16 * PAIRC];
  __shared__ __align__(16) __bf16 sA[16 * PAIRC];
  __shared__ __align__(16) __bf16 sT[16 * 64];
  __shared__ float mu_s[16], rs_s[16];
  const int tid = threadIdx.x;
  const int wave = tid >> 5;
  const int r0   = (tid & 16) ? 8 : 0;
  const int col  = tid & 15;
  const int ntiles = (LSEQ * LSEQ) / 16;
  int cur = 0;

  // prime buffer 0 with this block's first tile
  {
    size_t base = (size_t)blockIdx.x * (16 * PAIRC);
#pragma unroll
    for (int i = 0; i < 16; ++i) {
      int e = (i * 64 + tid) * 4;                  // 4 floats = 16 bytes
      async_copy_b128(&sX[0][e], pair + base + e);
    }
  }

  for (int t = blockIdx.x; t < ntiles; t += gridDim.x) {
    async_wait0();
    __syncthreads();
    // kick off the next tile into the other buffer (overlaps compute below)
    int tn = t + gridDim.x;
    if (tn < ntiles) {
      size_t base = (size_t)tn * (16 * PAIRC);
#pragma unroll
      for (int i = 0; i < 16; ++i) {
        int e = (i * 64 + tid) * 4;
        async_copy_b128(&sX[cur ^ 1][e], pair + base + e);
      }
    }
    const float* X = sX[cur];

    if (tid < 16) {
      float s = 0.f;
      for (int k = 0; k < PAIRC; ++k) s += X[tid * PAIRC + k];
      float mu = s / (float)PAIRC, v = 0.f;
      for (int k = 0; k < PAIRC; ++k) { float d = X[tid * PAIRC + k] - mu; v += d * d; }
      mu_s[tid] = mu;
      rs_s[tid] = rsqrtf(v / (float)PAIRC + 1e-5f);
    }
    __syncthreads();
    for (int idx = tid; idx < 16 * PAIRC; idx += 64) {
      int r = idx >> 8, k = idx & 255;
      sA[idx] = (__bf16)((X[idx] - mu_s[r]) * rs_s[r] * lng[k] + lnb[k]);
    }
    __syncthreads();

    // stage 1: (16x256) @ (256x64) + b1, gelu -> sT
    for (int ct = wave; ct < 4; ct += 2) {
      const int n0 = ct * 16;
      const __bf16* wt = w1p + (size_t)ct * 8 * 512;
      v8f acc = {};
#pragma unroll
      for (int ks = 0; ks < 8; ++ks) {
        v16bf a = load_a_bf(sA + ks * 32, PAIRC);
        v16bf b = load_b_packed(wt + (size_t)ks * 512);
        acc = wmma_bf16(a, b, acc);
      }
#pragma unroll
      for (int v = 0; v < 8; ++v)
        sT[(r0 + v) * 64 + n0 + col] = (__bf16)gelu_exact(acc[v] + b1[n0 + col]);
    }
    __syncthreads();

    // stage 2: (16x64) @ (64x16 padded) ; cols 0..7 are the H heads
    if (wave == 0) {
      v8f acc = {};
#pragma unroll
      for (int ks = 0; ks < 2; ++ks) {
        v16bf a = load_a_bf(sT + ks * 32, 64);
        v16bf b = load_b_packed(w2p + (size_t)ks * 512);
        acc = wmma_bf16(a, b, acc);
      }
      if (col < 8) {
        float bs = bias_scale[col];
#pragma unroll
        for (int v = 0; v < 8; ++v) {
          size_t gi = (size_t)col * LSEQ * LSEQ + (size_t)t * 16 + r0 + v;
          pb_out[gi] = bs * tanhf(acc[v] + b2[col]);
        }
      }
    }
    __syncthreads();
    cur ^= 1;
  }
}

// ---------------------------------------------------------------------------
// Flash attention per (head, 16-row i-tile), one wave per block.
// ---------------------------------------------------------------------------
__global__ __launch_bounds__(32)
void k_attention(const __bf16* __restrict__ qkv, const __bf16* __restrict__ vT,
                 const float* __restrict__ pb,
                 const float* __restrict__ gate_logits, int li,
                 __bf16* __restrict__ o) {
  const int head = blockIdx.x / (LSEQ / 16);
  const int i0   = (blockIdx.x % (LSEQ / 16)) * 16;
  const float gate  = 1.f / (1.f + expf(-gate_logits[li * NH + head]));
  const float scale = 0.17677669529663687f;   // HD^-0.5, HD=32
  const int r0  = (threadIdx.x & 16) ? 8 : 0;
  const int col = threadIdx.x & 15;
  const int qc = head * HD, kc = DM + head * HD;

  __shared__ __align__(16) __bf16 sP[16 * 32];
  const float* pbh = pb + (size_t)head * LSEQ * LSEQ;

  v16bf aq = load_a_bf(qkv + (size_t)i0 * (3 * DM) + qc, 3 * DM);
  float m[8], su[8];
  v8f acc0 = {}, acc1 = {};
#pragma unroll
  for (int v = 0; v < 8; ++v) { m[v] = -1e30f; su[v] = 0.f; }

  for (int j0 = 0; j0 < LSEQ; j0 += 32) {
    v16bf bk0 = load_a_bf(qkv + (size_t)j0 * (3 * DM) + kc, 3 * DM);
    v16bf bk1 = load_a_bf(qkv + (size_t)(j0 + 16) * (3 * DM) + kc, 3 * DM);
    v8f z = {};
    v8f s0 = wmma_bf16(aq, bk0, z);
    v8f s1 = wmma_bf16(aq, bk1, z);
#pragma unroll
    for (int v = 0; v < 8; ++v) {
      int row = i0 + r0 + v;
      float x0 = s0[v] * scale + gate * pbh[(size_t)row * LSEQ + j0 + col];
      float x1 = s1[v] * scale + gate * pbh[(size_t)row * LSEQ + j0 + 16 + col];
      float mx = hmax16(fmaxf(x0, x1));
      float mn = fmaxf(m[v], mx);
      float corr = expf(m[v] - mn);
      float p0 = expf(x0 - mn), p1 = expf(x1 - mn);
      su[v] = su[v] * corr + hsum16(p0 + p1);
      acc0[v] *= corr;
      acc1[v] *= corr;
      m[v] = mn;
      sP[(r0 + v) * 32 + col]      = (__bf16)p0;
      sP[(r0 + v) * 32 + 16 + col] = (__bf16)p1;
    }
    __syncthreads();
    v16bf ap  = load_a_bf(sP, 32);
    v16bf bv0 = load_a_bf(vT + (size_t)(head * HD) * LSEQ + j0, LSEQ);
    v16bf bv1 = load_a_bf(vT + (size_t)(head * HD + 16) * LSEQ + j0, LSEQ);
    acc0 = wmma_bf16(ap, bv0, acc0);
    acc1 = wmma_bf16(ap, bv1, acc1);
    __syncthreads();
  }
#pragma unroll
  for (int v = 0; v < 8; ++v) {
    int row = i0 + r0 + v;
    float inv = 1.f / su[v];
    o[(size_t)row * DM + head * HD + col]      = (__bf16)(acc0[v] * inv);
    o[(size_t)row * DM + head * HD + 16 + col] = (__bf16)(acc1[v] * inv);
  }
}

// ---------------------------------------------------------------------------
// Final LayerNorm per row, column mean, site MLP
// ---------------------------------------------------------------------------
__global__ __launch_bounds__(256)
void k_rowln(const float* __restrict__ X, const float* __restrict__ g,
             const float* __restrict__ b, float* __restrict__ Y, int K) {
  __shared__ float red[256];
  const int row = blockIdx.x, tid = threadIdx.x;
  const float* xr = X + (size_t)row * K;
  float s = 0.f;
  for (int k = tid; k < K; k += 256) s += xr[k];
  red[tid] = s; __syncthreads();
  for (int o2 = 128; o2 > 0; o2 >>= 1) { if (tid < o2) red[tid] += red[tid + o2]; __syncthreads(); }
  float mu = red[0] / (float)K; __syncthreads();
  float v = 0.f;
  for (int k = tid; k < K; k += 256) { float d = xr[k] - mu; v += d * d; }
  red[tid] = v; __syncthreads();
  for (int o2 = 128; o2 > 0; o2 >>= 1) { if (tid < o2) red[tid] += red[tid + o2]; __syncthreads(); }
  float rs = rsqrtf(red[0] / (float)K + 1e-5f);
  for (int k = tid; k < K; k += 256)
    Y[(size_t)row * K + k] = (xr[k] - mu) * rs * g[k] + b[k];
}

__global__ __launch_bounds__(256)
void k_colmean(const float* __restrict__ X, float* __restrict__ mean) {
  int c = threadIdx.x;
  float s = 0.f;
  for (int l = 0; l < LSEQ; ++l) s += X[(size_t)l * DM + c];
  mean[c] = s * (1.f / (float)LSEQ);
}

__global__ __launch_bounds__(128)
void k_site(const float* __restrict__ hf, const float* __restrict__ hmean,
            const int* __restrict__ pos, const int* __restrict__ wt,
            const int* __restrict__ mut,
            const float* __restrict__ w1, const float* __restrict__ b1,
            const float* __restrict__ w2, const float* __restrict__ b2,
            const float* __restrict__ w3, const float* __restrict__ b3,
            float* __restrict__ out) {
  __shared__ float feats[513], z1[128], z2[64], sc[20];
  const int site = blockIdx.x, tid = threadIdx.x;
  const int p = pos[site];
  for (int k = tid; k < DM; k += 128) {
    feats[k]      = hf[(size_t)p * DM + k];
    feats[DM + k] = hmean[k];
  }
  if (tid == 0) feats[512] = (float)p / (float)LSEQ;
  __syncthreads();
  { float a = 0.f; for (int k = 0; k < 513; ++k) a += feats[k] * w1[k * 128 + tid];
    z1[tid] = gelu_exact(a + b1[tid]); }
  __syncthreads();
  if (tid < 64) { float a = 0.f; for (int k = 0; k < 128; ++k) a += z1[k] * w2[k * 64 + tid];
    z2[tid] = gelu_exact(a + b2[tid]); }
  __syncthreads();
  if (tid < 20) { float a = 0.f; for (int k = 0; k < 64; ++k) a += z2[k] * w3[k * 20 + tid];
    sc[tid] = a + b3[tid]; }
  __syncthreads();
  if (tid == 0) out[site] = sc[mut[site]] - sc[wt[site]];
}

// ---------------------------------------------------------------------------
// Host launcher
// ---------------------------------------------------------------------------
extern "C" void kernel_launch(void* const* d_in, const int* in_sizes, int n_in,
                              void* d_out, int out_size, void* d_ws, size_t ws_size,
                              hipStream_t stream) {
  (void)in_sizes; (void)n_in; (void)out_size; (void)ws_size;
  enum { IN_SINGLE = 0, IN_PAIR, IN_POS, IN_WT, IN_MUT,
         IN_SPW, IN_SPB, IN_SNG, IN_SNB,
         IN_PBLNG, IN_PBLNB, IN_PBW1, IN_PBB1, IN_PBW2, IN_PBB2,
         IN_BSCALE, IN_GATE, IN_L0 = 17, IN_FNG = 65, IN_FNB = 66,
         IN_SW1 = 67, IN_SB1, IN_SW2, IN_SB2, IN_SW3, IN_SB3 };
  auto f32 = [&](int i) { return (const float*)d_in[i]; };
  auto i32 = [&](int i) { return (const int*)d_in[i]; };

  char* ws = (char*)d_ws;
  size_t off = 0;
  auto alloc = [&](size_t bytes) { size_t o = off; off += (bytes + 255) & ~(size_t)255; return o; };
  float*  pb    = (float*)(ws + alloc((size_t)NH * LSEQ * LSEQ * 4));
  float*  h     = (float*)(ws + alloc((size_t)LSEQ * DM * 4));
  __bf16* qkvb  = (__bf16*)(ws + alloc((size_t)LSEQ * 3 * DM * 2));
  __bf16* vtb   = (__bf16*)(ws + alloc((size_t)DM * LSEQ * 2));
  __bf16* ob    = (__bf16*)(ws + alloc((size_t)LSEQ * DM * 2));
  __bf16* tb    = (__bf16*)(ws + alloc((size_t)LSEQ * DFF * 2));
  float*  hf    = (float*)(ws + alloc((size_t)LSEQ * DM * 4));
  float*  hmean = (float*)(ws + alloc(DM * 4));
  __bf16* wsp   = (__bf16*)(ws + alloc((size_t)384 * DM * 2));
  __bf16* wp1   = (__bf16*)(ws + alloc((size_t)PAIRC * 64 * 2));
  __bf16* wp2   = (__bf16*)(ws + alloc((size_t)64 * 16 * 2));
  __bf16* wq[4]; __bf16* wo[4]; __bf16* w1[4]; __bf16* w2[4];
  for (int l = 0; l < 4; ++l) {
    wq[l] = (__bf16*)(ws + alloc((size_t)DM * 3 * DM * 2));
    wo[l] = (__bf16*)(ws + alloc((size_t)DM * DM * 2));
    w1[l] = (__bf16*)(ws + alloc((size_t)DM * DFF * 2));
    w2[l] = (__bf16*)(ws + alloc((size_t)DFF * DM * 2));
  }

  auto pack = [&](const float* s, __bf16* d, int K, int N) {
    k_packB<<<(K * N + 255) / 256, 256, 0, stream>>>(s, d, K, N);
  };
  pack(f32(IN_SPW), wsp, 384, DM);
  pack(f32(IN_PBW1), wp1, PAIRC, 64);
  k_packw2<<<4, 256, 0, stream>>>(f32(IN_PBW2), wp2);
  for (int l = 0; l < 4; ++l) {
    int b = IN_L0 + l * 12;
    pack(f32(b + 2), wq[l], DM, 3 * DM);
    pack(f32(b + 4), wo[l], DM, DM);
    pack(f32(b + 8), w1[l], DM, DFF);
    pack(f32(b + 10), w2[l], DFF, DM);
  }

  // h = LN(single @ Wsp + b)
  k_gemm<<<LSEQ / 16, 256, 0, stream>>>(
      f32(IN_SINGLE), nullptr, 384, DM, nullptr, nullptr, wsp, f32(IN_SPB),
      0, f32(IN_SNG), f32(IN_SNB), 0, h, nullptr, nullptr);

  // pair bias (dominant): persistent double-buffered async streaming
  k_pairbias<<<PB_GRID, 64, 0, stream>>>(
      f32(IN_PAIR), f32(IN_PBLNG), f32(IN_PBLNB), wp1, f32(IN_PBB1),
      wp2, f32(IN_PBB2), f32(IN_BSCALE), pb);

  for (int l = 0; l < 4; ++l) {
    int b = IN_L0 + l * 12;
    k_gemm<<<LSEQ / 16, 256, 0, stream>>>(
        h, nullptr, DM, 3 * DM, f32(b + 0), f32(b + 1), wq[l], f32(b + 3),
        0, nullptr, nullptr, 0, nullptr, qkvb, vtb);
    k_attention<<<NH * (LSEQ / 16), 32, 0, stream>>>(qkvb, vtb, pb, f32(IN_GATE), l, ob);
    k_gemm<<<LSEQ / 16, 256, 0, stream>>>(
        nullptr, ob, DM, DM, nullptr, nullptr, wo[l], f32(b + 5),
        0, nullptr, nullptr, 1, h, nullptr, nullptr);
    k_gemm<<<LSEQ / 16, 256, 0, stream>>>(
        h, nullptr, DM, DFF, f32(b + 6), f32(b + 7), w1[l], f32(b + 9),
        1, nullptr, nullptr, 0, nullptr, tb, nullptr);
    k_gemm<<<LSEQ / 16, 256, 0, stream>>>(
        nullptr, tb, DFF, DM, nullptr, nullptr, w2[l], f32(b + 11),
        0, nullptr, nullptr, 1, h, nullptr, nullptr);
  }

  k_rowln<<<LSEQ, 256, 0, stream>>>(h, f32(IN_FNG), f32(IN_FNB), hf, DM);
  k_colmean<<<1, 256, 0, stream>>>(hf, hmean);
  k_site<<<MMUT, 128, 0, stream>>>(
      hf, hmean, i32(IN_POS), i32(IN_WT), i32(IN_MUT),
      f32(IN_SW1), f32(IN_SB1), f32(IN_SW2), f32(IN_SB2), f32(IN_SW3), f32(IN_SB3),
      (float*)d_out);
}
